// CrossAttention_8435315769719
// MI455X (gfx1250) — matrix operand order repaired and checked by reading, not compiled, not run on hardware
//
#include <hip/hip_runtime.h>

typedef _Float16 f16;
typedef __attribute__((ext_vector_type(16))) _Float16 v16h;
typedef __attribute__((ext_vector_type(8)))  float    v8f;

#define HEADS  12
#define DHEAD  64
#define NQ     1024
#define NC     2048
#define DMODEL 768
#define FFI    3072
#define TLA    72   // LDS row stride (f16): 64 data + 8 pad = 144B, 16B-aligned

// ---- Tensor Data Mover availability ----------------------------------------
#if __has_builtin(__builtin_amdgcn_tensor_load_to_lds)
#define USE_TDM 1
#if __has_include(<hip/amd_detail/amd_gfx1250_TDM.h>)
#define TDM_6ARG 1   // therock/clang-23 lane: 6-arg builtin
#else
#define TDM_6ARG 0   // ROCm 7.2 / clang-22 lane: 5-arg builtin
#endif
#else
#define USE_TDM 0
#endif

#if USE_TDM
typedef __attribute__((ext_vector_type(4))) unsigned int u32x4;
typedef __attribute__((ext_vector_type(8))) int          i32x8;
typedef __attribute__((ext_vector_type(4))) int          i32x4;

// Issue one TDM descriptor: load a [tile_dim1=128 rows x tile_dim0=64 cols]
// f16 tile whose rows are K elements apart in global, into LDS with 16B of
// padding appended every 128B row (matches the TLA=72 fragment layout).
// D# bitfields per CDNA5 ISA 8.3/8.4 (group0: count/lds/global/type;
// group1: data_size, pad cfg, tensor dims, tile dims, row stride).
__device__ __forceinline__ void tdm_load_a_tile(const f16* gsrc,
                                                unsigned lds_addr,
                                                int K, int M) {
  unsigned long long ga = (unsigned long long)(size_t)gsrc;
  u32x4 g0;
  g0.x = 1u;                                    // count=1, is_restore=0
  g0.y = lds_addr;                              // LDS byte address
  g0.z = (unsigned)(ga & 0xffffffffu);          // global_addr[95:64 of grp]
  g0.w = (unsigned)((ga >> 32) & 0x1ffffffu) | (2u << 30);  // addr hi | type=2
  i32x8 g1;
  // data_size=1 (2B) | pad_enable | pad_interval=4 (32 DW) | pad_amount=3 (4 DW)
  g1[0] = (1 << 16) | (1 << 20) | (4 << 22) | (3 << 25);
  g1[1] = (K & 0xffff) << 16;                       // tensor_dim0[15:0]
  g1[2] = ((unsigned)K >> 16) | ((M & 0xffff) << 16); // dim0 hi | tensor_dim1 lo
  g1[3] = ((unsigned)M >> 16) | (64 << 16);         // dim1 hi | tile_dim0=64
  g1[4] = 128;                                      // tile_dim1=128, tile_dim2=0
  g1[5] = K;                                        // tensor_dim0_stride[31:0]
  g1[6] = 0;                                        // stride hi | dim1_stride lo
  g1[7] = 0;
  i32x4 gz = {0, 0, 0, 0};                          // groups 2/3: 2D tensor
#if TDM_6ARG
  i32x8 gz8 = {0, 0, 0, 0, 0, 0, 0, 0};
  __builtin_amdgcn_tensor_load_to_lds(g0, g1, gz, gz, gz8, 0);
#else
  __builtin_amdgcn_tensor_load_to_lds(g0, g1, gz, gz, 0);
#endif
}
#endif  // USE_TDM

// ---- fragment loaders -------------------------------------------------------
__device__ __forceinline__ v16h ld_b32(const f16* p) {  // 32 contiguous bytes
  v16h r;
  ((uint4*)&r)[0] = *(const uint4*)p;
  ((uint4*)&r)[1] = *(const uint4*)(p + 8);
  return r;
}
__device__ __forceinline__ v16h ld_a_split(const f16* p) {  // K split at +16
  v16h r;
  ((uint4*)&r)[0] = *(const uint4*)p;
  ((uint4*)&r)[1] = *(const uint4*)(p + 16);
  return r;
}

__device__ __forceinline__ float rmax16(float v) {
  v = fmaxf(v, __shfl_xor(v, 1));
  v = fmaxf(v, __shfl_xor(v, 2));
  v = fmaxf(v, __shfl_xor(v, 4));
  v = fmaxf(v, __shfl_xor(v, 8));
  return v;
}
__device__ __forceinline__ float rsum16(float v) {
  v += __shfl_xor(v, 1);
  v += __shfl_xor(v, 2);
  v += __shfl_xor(v, 4);
  v += __shfl_xor(v, 8);
  return v;
}

// ---- LayerNorm (row of 768) -> f16 -----------------------------------------
__global__ void __launch_bounds__(256)
ln_f16(const float* __restrict__ in, const float* __restrict__ gamma,
       f16* __restrict__ out) {
  __shared__ float s1[256], s2[256];
  const int t = threadIdx.x;
  const size_t base = (size_t)blockIdx.x * DMODEL;
  float v0 = in[base + t], v1 = in[base + t + 256], v2 = in[base + t + 512];
  s1[t] = v0 + v1 + v2;
  s2[t] = v0 * v0 + v1 * v1 + v2 * v2;
  __syncthreads();
  for (int off = 128; off > 0; off >>= 1) {
    if (t < off) { s1[t] += s1[t + off]; s2[t] += s2[t + off]; }
    __syncthreads();
  }
  const float mu  = s1[0] * (1.f / DMODEL);
  const float var = s2[0] * (1.f / DMODEL) - mu * mu;
  const float rs  = rsqrtf(var + 1e-5f);
  out[base + t]       = (f16)((v0 - mu) * rs * gamma[t]);
  out[base + t + 256] = (f16)((v1 - mu) * rs * gamma[t + 256]);
  out[base + t + 512] = (f16)((v2 - mu) * rs * gamma[t + 512]);
}

// ---- TDM/register double-buffered WMMA GEMM --------------------------------
// C[M,N] = A_f16[M,K] * B_f32[K,*]; 256 threads = 8 waves (4x2),
// macro tile 128x64, wave tile 32x32, K-stage 64 (2 WMMA sub-steps).
// A tiles: TDM DMA into ping-pong LDS buffers (ASYNC overlap, padded rows),
//          or register-staged fallback when the TDM builtin is absent.
// B tiles: fp32 loads, converted and committed as packed b128 LDS stores.
// MODE 1: q proj -> [b,h,n,64] f16, *0.125
// MODE 2: kv -> k [b,j,64] f16 ; v stored transposed [b,64,j] f16
// MODE 3: store f32 | MODE 4: += f32 | MODE 5: fused SwiGLU -> f16
template<int MODE>
__global__ void __launch_bounds__(256)
gemm_f16(const f16* __restrict__ A, const float* __restrict__ B,
         f16* __restrict__ o16, f16* __restrict__ o16b,
         float* __restrict__ of32, int M, int N, int K, int ldB) {
#if USE_TDM
  __shared__ __align__(16) f16 As[2][128 * TLA];   // ping-pong DMA targets
#else
  __shared__ __align__(16) f16 As[1][128 * TLA];
#endif
  __shared__ __align__(16) f16 Bs[64 * TLA];
  __shared__ __align__(16) f16 Bs2[(MODE == 5) ? 64 * TLA : 16];
  const int t = threadIdx.x;
  const int lane = t & 31, w = t >> 5;
  const int wm = w >> 1, wn = w & 1;
  const int half = lane >> 4, l15 = lane & 15;
  const int m0 = blockIdx.x * 128, n0 = blockIdx.y * 64;

  const int bn = t & 63;            // B column n within tile
  const int bk = (t >> 6) << 4;     // B k-base (16 consecutive k per thread)
#if !USE_TDM
  const int ar = t >> 1;            // A row (2 threads per row)
  const int ac = (t & 1) * 32;      // A col 0 or 32
  uint4 ra[2];
#endif

  const v8f zero = {0.f, 0.f, 0.f, 0.f, 0.f, 0.f, 0.f, 0.f};
  v8f acc[2][2]  = {{zero, zero}, {zero, zero}};
  v8f accg[2][2] = {{zero, zero}, {zero, zero}};

  float rb[16], rb2[16];
  const int KT = K >> 6;

  // prologue: stage 0 in flight
#if USE_TDM
  const unsigned ldsA0 = (unsigned)(size_t)&As[0][0];
  const unsigned ldsA1 = (unsigned)(size_t)&As[1][0];
  if (w == 0) tdm_load_a_tile(A + (size_t)m0 * K, ldsA0, K, M);
#else
  {
    const f16* ap = A + (size_t)(m0 + ar) * K + ac;
    ra[0] = *(const uint4*)ap;
    ra[1] = *(const uint4*)(ap + 8);
  }
#endif
#pragma unroll
  for (int i = 0; i < 16; ++i) {
    rb[i] = B[(size_t)(bk + i) * ldB + n0 + bn];
    if (MODE == 5) rb2[i] = B[(size_t)(bk + i) * ldB + FFI + n0 + bn];
  }

  for (int kt = 0; kt < KT; ++kt) {
#if USE_TDM
    const f16* Acur = &As[kt & 1][0];
#else
    const f16* Acur = &As[0][0];
#endif
    __syncthreads();  // prior-stage LDS readers done
    // commit B stage regs -> LDS (transposed, packed b128 stores)
    {
      __align__(16) f16 hb[16];
#pragma unroll
      for (int i = 0; i < 16; ++i) hb[i] = (f16)rb[i];
      *(uint4*)&Bs[bn * TLA + bk]     = *(uint4*)&hb[0];
      *(uint4*)&Bs[bn * TLA + bk + 8] = *(uint4*)&hb[8];
      if (MODE == 5) {
#pragma unroll
        for (int i = 0; i < 16; ++i) hb[i] = (f16)rb2[i];
        *(uint4*)&Bs2[bn * TLA + bk]     = *(uint4*)&hb[0];
        *(uint4*)&Bs2[bn * TLA + bk + 8] = *(uint4*)&hb[8];
      }
    }
#if USE_TDM
    // issue next DMA into the other buffer (its readers finished before the
    // barrier above), then wait for the current tile; TDM completes in-order,
    // so tensorcnt<=1 with two outstanding means tile kt has landed.
    if (w == 0) {
      if (kt + 1 < KT) {
        tdm_load_a_tile(A + (size_t)m0 * K + ((size_t)(kt + 1) << 6),
                        (kt & 1) ? ldsA0 : ldsA1, K, M);
        __builtin_amdgcn_s_wait_tensorcnt(1);
      } else {
        __builtin_amdgcn_s_wait_tensorcnt(0);
      }
    }
#else
    *(uint4*)&As[0][ar * TLA + ac]     = ra[0];
    *(uint4*)&As[0][ar * TLA + ac + 8] = ra[1];
#endif
    __syncthreads();  // tile kt (A via DMA/regs, B via stores) visible

    // fetch next B stage into regs (overlaps the WMMAs below)
    if (kt + 1 < KT) {
      const int k0 = (kt + 1) << 6;
#if !USE_TDM
      const f16* ap = A + (size_t)(m0 + ar) * K + k0 + ac;
      ra[0] = *(const uint4*)ap;
      ra[1] = *(const uint4*)(ap + 8);
#endif
#pragma unroll
      for (int i = 0; i < 16; ++i) {
        rb[i] = B[(size_t)(k0 + bk + i) * ldB + n0 + bn];
        if (MODE == 5) rb2[i] = B[(size_t)(k0 + bk + i) * ldB + FFI + n0 + bn];
      }
      __builtin_prefetch(&B[(size_t)(k0 + 64 + bk) * ldB + n0 + bn], 0, 0);
    }

    // compute current stage: 2 K sub-steps of 32
#pragma unroll
    for (int ks = 0; ks < 2; ++ks) {
      v16h a[2], b[2], b2[2];
#pragma unroll
      for (int i = 0; i < 2; ++i)
        a[i] = ld_b32(&Acur[(wm * 32 + i * 16 + l15) * TLA + ks * 32 + half * 16]);
#pragma unroll
      for (int j = 0; j < 2; ++j) {
        b[j] = ld_b32(&Bs[(wn * 32 + j * 16 + l15) * TLA + ks * 32 + half * 16]);
        if (MODE == 5)
          b2[j] = ld_b32(&Bs2[(wn * 32 + j * 16 + l15) * TLA + ks * 32 + half * 16]);
      }
#pragma unroll
      for (int i = 0; i < 2; ++i)
#pragma unroll
        for (int j = 0; j < 2; ++j) {
          acc[i][j] = __builtin_amdgcn_wmma_f32_16x16x32_f16(
              false, a[i], false, b[j], (short)0, acc[i][j], false, false);
          if (MODE == 5)
            accg[i][j] = __builtin_amdgcn_wmma_f32_16x16x32_f16(
                false, a[i], false, b2[j], (short)0, accg[i][j], false, false);
        }
    }
  }

#pragma unroll
  for (int i = 0; i < 2; ++i)
#pragma unroll
    for (int j = 0; j < 2; ++j)
#pragma unroll
      for (int r = 0; r < 8; ++r) {
        int row = m0 + wm * 32 + i * 16 + r + half * 8;
        int col = n0 + wn * 32 + j * 16 + l15;
        float v = acc[i][j][r];
        if (MODE == 1) {
          int b_ = row >> 10, n_ = row & 1023;
          int h_ = col >> 6, d_ = col & 63;
          o16[(((size_t)b_ * HEADS + h_) * NQ + n_) * DHEAD + d_] =
              (f16)(v * 0.125f);
        } else if (MODE == 2) {
          int b_ = row >> 11, jj = row & 2047;
          if (col < DHEAD)
            o16[((size_t)b_ * NC + jj) * DHEAD + col] = (f16)v;
          else
            o16b[((size_t)b_ * DHEAD + (col - DHEAD)) * NC + jj] = (f16)v;
        } else if (MODE == 3) {
          of32[(size_t)row * N + col] = v;
        } else if (MODE == 4) {
          float* p = of32 + (size_t)row * N + col;
          *p += v;
        } else if (MODE == 5) {
          float g = accg[i][j][r];
          float act = g / (1.f + __expf(-g));
          o16[(size_t)row * N + col] = (f16)(v * act);
        }
      }
}

// ---- flash-attention: grid (8 q-blocks, 48 b*h), 8 waves x 16 q-rows -------
__global__ void __launch_bounds__(256)
attn_f16(const f16* __restrict__ qb, const f16* __restrict__ kb,
         const f16* __restrict__ vt, f16* __restrict__ ob) {
  __shared__ __align__(16) f16 Pb[8][16][72];  // per-wave P tile [16 q][64 j]
  const int t = threadIdx.x;
  const int lane = t & 31, w = t >> 5;
  const int half = lane >> 4, l15 = lane & 15;
  const int bh = blockIdx.y;
  const int b = bh / HEADS, h = bh % HEADS;
  const int q0 = blockIdx.x * 128 + w * 16;

  const f16* qp = qb + ((size_t)bh * NQ + q0) * DHEAD;  // [16][64]
  const f16* kp = kb + (size_t)b * NC * DHEAD;          // [j][64]
  const f16* vp = vt + (size_t)b * DHEAD * NC;          // [d][j] (v^T)

  v16h qa[2];
#pragma unroll
  for (int ks = 0; ks < 2; ++ks)
    qa[ks] = ld_b32(qp + (size_t)l15 * DHEAD + ks * 32 + half * 16);

  const v8f zero = {0.f, 0.f, 0.f, 0.f, 0.f, 0.f, 0.f, 0.f};
  v8f acc[4] = {zero, zero, zero, zero};  // out tile [16][64]
  float mrun[8], lrun[8];
#pragma unroll
  for (int r = 0; r < 8; ++r) { mrun[r] = -1e30f; lrun[r] = 0.f; }

  for (int j0 = 0; j0 < NC; j0 += 64) {
    v8f S[4];
#pragma unroll
    for (int nt = 0; nt < 4; ++nt) {
      v8f s = zero;
#pragma unroll
      for (int ks = 0; ks < 2; ++ks) {
        v16h kf = ld_b32(kp + (size_t)(j0 + nt * 16 + l15) * DHEAD +
                         ks * 32 + half * 16);
        s = __builtin_amdgcn_wmma_f32_16x16x32_f16(false, qa[ks], false, kf,
                                                   (short)0, s, false, false);
      }
      S[nt] = s;
    }
#pragma unroll
    for (int r = 0; r < 8; ++r) {
      float mx = fmaxf(fmaxf(S[0][r], S[1][r]), fmaxf(S[2][r], S[3][r]));
      mx = rmax16(mx);
      float mnew = fmaxf(mrun[r], mx);
      float alpha = __expf(mrun[r] - mnew);
      mrun[r] = mnew;
      float ps = 0.f;
#pragma unroll
      for (int nt = 0; nt < 4; ++nt) {
        float p = __expf(S[nt][r] - mnew);
        ps += p;
        Pb[w][r + half * 8][nt * 16 + l15] = (f16)p;
      }
      ps = rsum16(ps);
      lrun[r] = lrun[r] * alpha + ps;
#pragma unroll
      for (int dt = 0; dt < 4; ++dt) acc[dt][r] = acc[dt][r] * alpha;
    }
    asm volatile("s_wait_dscnt 0x0" ::: "memory");
    v16h pf[2];
#pragma unroll
    for (int ks = 0; ks < 2; ++ks)
      pf[ks] = ld_b32(&Pb[w][l15][ks * 32 + half * 16]);
#pragma unroll
    for (int dt = 0; dt < 4; ++dt) {
#pragma unroll
      for (int ks = 0; ks < 2; ++ks) {
        v16h vf = ld_b32(vp + (size_t)(dt * 16 + l15) * NC + j0 +
                         ks * 32 + half * 16);
        acc[dt] = __builtin_amdgcn_wmma_f32_16x16x32_f16(
            false, pf[ks], false, vf, (short)0, acc[dt], false, false);
      }
    }
  }
#pragma unroll
  for (int dt = 0; dt < 4; ++dt)
#pragma unroll
    for (int r = 0; r < 8; ++r) {
      int n = q0 + r + half * 8;
      int d = dt * 16 + l15;
      ob[((size_t)(b * NQ + n)) * DMODEL + h * DHEAD + d] =
          (f16)(acc[dt][r] / lrun[r]);
    }
}

// ---- host-side orchestration ------------------------------------------------
extern "C" void kernel_launch(void* const* d_in, const int* in_sizes, int n_in,
                              void* d_out, int out_size, void* d_ws,
                              size_t ws_size, hipStream_t stream) {
  const float* x    = (const float*)d_in[0];
  const float* ctx  = (const float*)d_in[1];
  const float* gx   = (const float*)d_in[2];
  const float* gc   = (const float*)d_in[3];
  const float* Wq   = (const float*)d_in[4];
  const float* Wkv  = (const float*)d_in[5];
  const float* Wo   = (const float*)d_in[6];
  const float* Wff1 = (const float*)d_in[7];
  const float* Wff2 = (const float*)d_in[8];
  float* out = (float*)d_out;

  char* ws = (char*)d_ws;
  f16* xn  = (f16*)(ws + 0);         // 4096x768
  f16* cn  = (f16*)(ws + 6291456);   // 8192x768
  f16* qB  = (f16*)(ws + 18874368);  // [b,h,1024,64]
  f16* kB  = (f16*)(ws + 25165824);  // [b,2048,64]
  f16* vT  = (f16*)(ws + 26214400);  // [b,64,2048]
  f16* ao  = (f16*)(ws + 27262976);  // 4096x768
  f16* ffa = (f16*)(ws + 33554432);  // 4096x3072   (total 58,720,256 B)

  ln_f16<<<dim3(4096), dim3(256), 0, stream>>>(x, gx, xn);
  ln_f16<<<dim3(8192), dim3(256), 0, stream>>>(ctx, gc, cn);
  gemm_f16<1><<<dim3(32, 12), dim3(256), 0, stream>>>(
      xn, Wq, qB, nullptr, nullptr, 4096, DMODEL, DMODEL, DMODEL);
  gemm_f16<2><<<dim3(64, 2), dim3(256), 0, stream>>>(
      cn, Wkv, kB, vT, nullptr, 8192, 2 * DHEAD, DMODEL, 2 * DHEAD);
  attn_f16<<<dim3(8, 48), dim3(256), 0, stream>>>(qB, kB, vT, ao);
  gemm_f16<3><<<dim3(32, 12), dim3(256), 0, stream>>>(
      ao, Wo, nullptr, nullptr, out, 4096, DMODEL, DMODEL, DMODEL);
  gemm_f16<5><<<dim3(32, 48), dim3(256), 0, stream>>>(
      xn, Wff1, ffa, nullptr, nullptr, 4096, FFI, DMODEL, 2 * FFI);
  gemm_f16<4><<<dim3(32, 12), dim3(256), 0, stream>>>(
      ffa, Wff2, nullptr, nullptr, out, 4096, DMODEL, FFI, DMODEL);

  (void)in_sizes; (void)n_in; (void)out_size; (void)ws_size;
}
